// RotorQuantAttention_18605798327028
// MI455X (gfx1250) — matrix operand order repaired and checked by reading, not compiled, not run on hardware
//
#include <hip/hip_runtime.h>
#include <hip/hip_bf16.h>
#include <math.h>

#define HEADS 32
#define DH    64
#define DM    2048
#define BATCH 2
#define SEQ   1024
#define NROWS (BATCH*SEQ)   // 2048 token rows
#define NG    22            // rotor groups (66 padded dims / 3)

typedef __attribute__((ext_vector_type(16))) __bf16 bf16x16;
typedef __attribute__((ext_vector_type(8)))  float  f32x8;

union Frag16 { bf16x16 v; uint4 q[2]; };

static __device__ __forceinline__ f32x8 wmma_bf16(const Frag16& a, const Frag16& b, f32x8 c) {
  // D = A(16x32) * B(32x16) + C, f32 accumulate
  return __builtin_amdgcn_wmma_f32_16x16x32_bf16(false, a.v, false, b.v, (short)0, c, false, false);
}

// ---------------------------------------------------------------------------
// Weight transpose + f32 -> bf16 convert.  W is (K,N) row-major (x @ W),
// output is (N,K) row-major bf16 so GEMM B-fragments read contiguous K.
// ---------------------------------------------------------------------------
__global__ __launch_bounds__(256) void k_wtrans(const float* __restrict__ W0, const float* __restrict__ W1,
                                                const float* __restrict__ W2, const float* __restrict__ W3,
                                                __bf16* __restrict__ WT3, __bf16* __restrict__ WoT)
{
  const float* W = (blockIdx.z==0)?W0:(blockIdx.z==1)?W1:(blockIdx.z==2)?W2:W3;
  __bf16* out = (blockIdx.z<3) ? (WT3 + (size_t)blockIdx.z*DM*DM) : WoT;
  __shared__ float tile[32][33];
  const int tx = threadIdx.x, ty = threadIdx.y;
  const int k0 = blockIdx.y*32, n0 = blockIdx.x*32;
  #pragma unroll
  for (int i=0;i<32;i+=8) tile[ty+i][tx] = W[(size_t)(k0+ty+i)*DM + n0+tx];
  __syncthreads();
  #pragma unroll
  for (int i=0;i<32;i+=8) out[(size_t)(n0+ty+i)*DM + k0+tx] = (__bf16)tile[tx][ty+i];
}

// ---------------------------------------------------------------------------
// Elementwise f32 -> bf16
// ---------------------------------------------------------------------------
__global__ __launch_bounds__(256) void k_cvt(const float* __restrict__ src, __bf16* __restrict__ dst, int n4)
{
  int i = blockIdx.x*256 + threadIdx.x;
  if (i < n4) {
    float4 v = ((const float4*)src)[i];
    __bf16* d = dst + (size_t)i*4;
    d[0]=(__bf16)v.x; d[1]=(__bf16)v.y; d[2]=(__bf16)v.z; d[3]=(__bf16)v.w;
  }
}

// ---------------------------------------------------------------------------
// GEMM: D(M,N) f32 = A(M,K) bf16 @ BT(N,K)^T bf16.
// Block = 8 waves = 128 rows x 64 cols; wave tile 16x64 (4 accumulators).
// No LDS: fragments are direct b128 global loads (operands L2-resident).
// 2-deep software pipeline over K; small tile keeps both stages (~140 VGPRs)
// live so WMMA groups overlap the next group's 10 in-flight loads.
// Addressing: uniform base + 32-bit element offsets (saddr+voffset form).
// ---------------------------------------------------------------------------
__global__ __launch_bounds__(256) void k_gemm(const __bf16* __restrict__ A, const __bf16* __restrict__ BT,
                                              float* __restrict__ D, int M, int N, int K)
{
  const int wave = threadIdx.x >> 5;
  const int lane = threadIdx.x & 31;
  const int l15  = lane & 15;
  const int hi   = lane >> 4;
  const int m0 = blockIdx.y*128 + wave*16;
  const int n0 = blockIdx.x*64;

  f32x8 acc[4];
  #pragma unroll
  for (int j=0;j<4;j++)
    #pragma unroll
    for (int g=0;g<8;g++) acc[j][g] = 0.f;

  const unsigned aoff0 = (unsigned)(m0 + l15)*K + hi*16;  // A 16x32: contiguous K per lane
  const unsigned boff0 = (unsigned)(n0 + l15)*K + hi*16;  // B 32x16: contiguous K per lane

  auto loadA = [&](Frag16& f, int kc) {
    const __bf16* p = A + (aoff0 + (unsigned)kc);
    f.q[0] = *(const uint4*)p;
    f.q[1] = *(const uint4*)(p + 8);
  };
  auto loadB = [&](Frag16* f, int kc) {
    #pragma unroll
    for (int j=0;j<4;j++) {
      const __bf16* p = BT + (boff0 + (unsigned)(j*16*K + kc));
      f[j].q[0] = *(const uint4*)p;
      f[j].q[1] = *(const uint4*)(p + 8);
    }
  };

  Frag16 af[2];
  Frag16 bf[2][4];
  loadA(af[0], 0);
  loadB(bf[0], 0);
  for (int kc = 0; kc < K; kc += 64) {
    loadA(af[1], kc + 32);
    loadB(bf[1], kc + 32);
    #pragma unroll
    for (int j=0;j<4;j++) acc[j] = wmma_bf16(af[0], bf[0][j], acc[j]);
    if (kc + 64 < K) {
      loadA(af[0], kc + 64);
      loadB(bf[0], kc + 64);
    }
    #pragma unroll
    for (int j=0;j<4;j++) acc[j] = wmma_bf16(af[1], bf[1][j], acc[j]);
  }

  #pragma unroll
  for (int j=0;j<4;j++)
    #pragma unroll
    for (int g=0;g<8;g++)
      D[(size_t)(m0 + g + hi*8)*N + n0 + j*16 + l15] = acc[j][g];
}

// ---------------------------------------------------------------------------
// RMSNorm over a 2048-wide row.  One block per row.
// dstB != nullptr -> write bf16 (DM stride), else write f32 (ldDstF stride).
// ---------------------------------------------------------------------------
__global__ __launch_bounds__(256) void k_rmsnorm(const float* __restrict__ src, const float* __restrict__ w,
                                                 float* __restrict__ dstF, __bf16* __restrict__ dstB,
                                                 int ldSrc, int ldDstF)
{
  const int row = blockIdx.x;
  const float* r = src + (size_t)row*ldSrc;
  float vals[8];
  float ss = 0.f;
  #pragma unroll
  for (int i=0;i<8;i++) { float v = r[threadIdx.x + i*256]; vals[i]=v; ss += v*v; }
  #pragma unroll
  for (int off=16; off>=1; off>>=1) ss += __shfl_xor(ss, off, 32);
  __shared__ float red[8];
  if ((threadIdx.x & 31)==0) red[threadIdx.x>>5] = ss;
  __syncthreads();
  float tot = 0.f;
  #pragma unroll
  for (int i=0;i<8;i++) tot += red[i];
  const float scale = rsqrtf(tot * (1.0f/DM) + 1e-6f);
  #pragma unroll
  for (int i=0;i<8;i++) {
    int c = threadIdx.x + i*256;
    float o = vals[i]*scale*w[c];
    if (dstB) dstB[(size_t)row*DM + c] = (__bf16)o;
    else      dstF[(size_t)row*ldDstF + c] = o;
  }
}

// ---------------------------------------------------------------------------
// RotorQuant roundtrip.  One thread per (b,h,s) 64-vector.
// Uniform codebook C => nearest level == clamped round((v-C0)/step).
// outRow != nullptr -> bf16 (B,S,DM) row-major   (K path, feeds QK^T B-frags)
// outT   != nullptr -> bf16 (B,H,Dh,S) transposed (V path, feeds PV  B-frags)
// ---------------------------------------------------------------------------
__global__ __launch_bounds__(256) void k_quant(const float* __restrict__ src, int ldSrc,
                                               const float* __restrict__ Mrot, const float* __restrict__ Mtr,
                                               const float* __restrict__ C,
                                               __bf16* __restrict__ outRow, __bf16* __restrict__ outT)
{
  __shared__ float sM[NG*9], sMt[NG*9];
  for (int i=threadIdx.x; i<NG*9; i+=256) { sM[i]=Mrot[i]; sMt[i]=Mtr[i]; }
  __syncthreads();

  const int t  = blockIdx.x*256 + threadIdx.x;   // (b*HEADS + h)*SEQ + s
  const int s  = t & (SEQ-1);
  const int bh = t >> 10;
  const int h  = bh & (HEADS-1);
  const int b  = bh >> 5;

  const float* x = src + (size_t)(b*SEQ + s)*ldSrc + h*DH;
  float xv[66];
  float ss = 0.f;
  #pragma unroll
  for (int i=0;i<DH;i++) { float v = x[i]; xv[i]=v; ss += v*v; }
  xv[64]=0.f; xv[65]=0.f;
  float nrm = fmaxf(sqrtf(ss), 1e-8f);
  const float inv = 1.f/nrm;

  const float c0    = C[0];
  const float step  = (C[7]-C[0]) * (1.f/7.f);
  const float istep = 1.f/step;

  float out[66];
  #pragma unroll
  for (int g=0; g<NG; g++) {
    const float* m  = sM  + g*9;
    const float* mt = sMt + g*9;
    float a0 = xv[3*g]*inv, a1 = xv[3*g+1]*inv, a2 = xv[3*g+2]*inv;
    float r0 = a0*m[0] + a1*m[3] + a2*m[6];
    float r1 = a0*m[1] + a1*m[4] + a2*m[7];
    float r2 = a0*m[2] + a1*m[5] + a2*m[8];
    float d0 = c0 + fminf(fmaxf(roundf((r0-c0)*istep),0.f),7.f)*step;
    float d1 = c0 + fminf(fmaxf(roundf((r1-c0)*istep),0.f),7.f)*step;
    float d2 = c0 + fminf(fmaxf(roundf((r2-c0)*istep),0.f),7.f)*step;
    out[3*g]   = (d0*mt[0] + d1*mt[3] + d2*mt[6])*nrm;
    out[3*g+1] = (d0*mt[1] + d1*mt[4] + d2*mt[7])*nrm;
    out[3*g+2] = (d0*mt[2] + d1*mt[5] + d2*mt[8])*nrm;
  }

  if (outRow) {
    __bf16* o = outRow + (size_t)(b*SEQ + s)*DM + h*DH;
    #pragma unroll
    for (int i=0;i<DH;i++) o[i] = (__bf16)out[i];
  } else {
    __bf16* o = outT + (size_t)(b*HEADS + h)*DH*SEQ + s;
    #pragma unroll
    for (int i=0;i<DH;i++) o[(size_t)i*SEQ] = (__bf16)out[i];
  }
}

// ---------------------------------------------------------------------------
// Flash attention for one (b,h): block = 8 waves = 128 query rows,
// inner loop over 32-key blocks.  All K and V fragments for the block are
// issued at the top of the iteration: score WMMAs wait only for the K loads
// (V's 8 loads stay in flight), and the V loads hide under the softmax VALU
// chain so the PV WMMAs find operands resident.  Next block's K/V lines are
// prefetched (global_prefetch_b8) one iteration ahead.
// ---------------------------------------------------------------------------
__global__ __launch_bounds__(256) void k_attn(const __bf16* __restrict__ qb, const __bf16* __restrict__ kb,
                                              const __bf16* __restrict__ vtb, __bf16* __restrict__ ob)
{
  const int bh   = blockIdx.x;           // 0..63
  const int b    = bh >> 5, h = bh & 31;
  const int wave = threadIdx.x >> 5;
  const int lane = threadIdx.x & 31;
  const int l15  = lane & 15;
  const int hi   = lane >> 4;
  const int q0   = blockIdx.y*128 + wave*16;

  __shared__ __align__(16) __bf16 plds[8][16][32];  // per-wave P tile, 8 KB

  Frag16 qa[2];                                     // Q rows as A-frags, dh chunks of 32
  {
    const __bf16* qrow = qb + (size_t)(b*SEQ + q0 + l15)*DM + h*DH;
    #pragma unroll
    for (int c=0;c<2;c++) {
      const __bf16* p = qrow + c*32 + hi*16;
      qa[c].q[0] = *(const uint4*)p;
      qa[c].q[1] = *(const uint4*)(p + 8);
    }
  }

  // uniform bases + 32-bit per-lane offsets (saddr + voffset form)
  const unsigned koff0 = (unsigned)(b*SEQ + l15)*DM + h*DH + hi*16;  // + key*DM + chunk*32
  const unsigned voff0 = (unsigned)(bh*DH + l15)*SEQ + hi*16;        // + slab*16*SEQ + key

  float m8[8], l8[8];
  f32x8 o[4];
  #pragma unroll
  for (int g=0;g<8;g++) { m8[g] = -1e30f; l8[g] = 0.f; }
  #pragma unroll
  for (int s4=0;s4<4;s4++)
    #pragma unroll
    for (int g=0;g<8;g++) o[s4][g] = 0.f;

  for (int kb0=0; kb0<SEQ; kb0+=32) {
    // ---- issue all operand loads for this key block up front ----
    Frag16 kf[2][2];                       // [key tile][dh chunk]
    #pragma unroll
    for (int t=0;t<2;t++)
      #pragma unroll
      for (int cc=0;cc<2;cc++) {
        const __bf16* p = kb + (koff0 + (unsigned)((kb0 + t*16)*DM + cc*32));
        kf[t][cc].q[0] = *(const uint4*)p;
        kf[t][cc].q[1] = *(const uint4*)(p + 8);
      }
    Frag16 vf[4];                          // [dh slab]
    #pragma unroll
    for (int s4=0;s4<4;s4++) {
      const __bf16* p = vtb + (voff0 + (unsigned)(s4*16*SEQ + kb0));
      vf[s4].q[0] = *(const uint4*)p;
      vf[s4].q[1] = *(const uint4*)(p + 8);
    }
    // warm WGP cache for the next key block while this one computes
    if (kb0 + 32 < SEQ) {
      __builtin_prefetch(kb + (koff0 + (unsigned)((kb0 + 32 + 16*hi)*DM)), 0, 3);
      __builtin_prefetch(vtb + (voff0 + (unsigned)(hi*2*16*SEQ + kb0 + 32)), 0, 3);
    }

    // ---- scores: two 16x16 tiles (keys kb0..+15, kb0+16..+31) ----
    f32x8 st[2];
    #pragma unroll
    for (int t=0;t<2;t++) {
      f32x8 c;
      #pragma unroll
      for (int g=0;g<8;g++) c[g] = 0.f;
      c = wmma_bf16(qa[0], kf[t][0], c);
      c = wmma_bf16(qa[1], kf[t][1], c);
      #pragma unroll
      for (int g=0;g<8;g++) st[t][g] = c[g]*0.125f;  // 1/sqrt(64)
    }

    // ---- online softmax over the 32 new columns ----
    float corr[8];
    #pragma unroll
    for (int g=0; g<8; g++) {
      float v = fmaxf(st[0][g], st[1][g]);
      #pragma unroll
      for (int off=8; off>=1; off>>=1) v = fmaxf(v, __shfl_xor(v, off, 16));
      float mn = fmaxf(m8[g], v);
      corr[g] = __expf(m8[g]-mn);
      m8[g] = mn;
      float p0 = __expf(st[0][g]-mn); st[0][g] = p0;
      float p1 = __expf(st[1][g]-mn); st[1][g] = p1;
      float rs = p0 + p1;
      #pragma unroll
      for (int off=8; off>=1; off>>=1) rs += __shfl_xor(rs, off, 16);
      l8[g] = l8[g]*corr[g] + rs;
    }

    // ---- P: D-layout -> LDS row-major 16x32 -> A-frag (same-wave, in order) ----
    #pragma unroll
    for (int t=0;t<2;t++)
      #pragma unroll
      for (int g=0;g<8;g++)
        plds[wave][g + hi*8][t*16 + l15] = (__bf16)st[t][g];
    Frag16 pa;
    {
      const __bf16* pp = &plds[wave][l15][hi*16];
      pa.q[0] = *(const uint4*)pp;
      pa.q[1] = *(const uint4*)(pp + 8);
    }

    // ---- O = O*corr + P @ V (4 dh slabs of 16) ----
    #pragma unroll
    for (int s4=0;s4<4;s4++) {
      #pragma unroll
      for (int g=0;g<8;g++) o[s4][g] *= corr[g];
      o[s4] = wmma_bf16(pa, vf[s4], o[s4]);
    }
  }

  #pragma unroll
  for (int s4=0;s4<4;s4++)
    #pragma unroll
    for (int g=0;g<8;g++) {
      float val = o[s4][g] / l8[g];
      ob[(size_t)(b*SEQ + q0 + g + hi*8)*DM + h*DH + s4*16 + l15] = (__bf16)val;
    }
}

// ---------------------------------------------------------------------------
extern "C" void kernel_launch(void* const* d_in, const int* in_sizes, int n_in,
                              void* d_out, int out_size, void* d_ws, size_t ws_size,
                              hipStream_t stream)
{
  const float* x    = (const float*)d_in[0];
  const float* Wq   = (const float*)d_in[1];
  const float* Wk   = (const float*)d_in[2];
  const float* Wv   = (const float*)d_in[3];
  const float* Wo   = (const float*)d_in[4];
  const float* qn_w = (const float*)d_in[5];
  const float* kn_w = (const float*)d_in[6];
  const float* Mk   = (const float*)d_in[7];
  const float* Mtk  = (const float*)d_in[8];
  const float* Ck   = (const float*)d_in[9];
  const float* Mv   = (const float*)d_in[10];
  const float* Mtv  = (const float*)d_in[11];
  const float* Cv   = (const float*)d_in[12];

  char* ws = (char*)d_ws;
  size_t off = 0;
  auto take = [&](size_t bytes) { size_t o = off; off += (bytes + 255) & ~(size_t)255; return o; };
  __bf16* WT3 = (__bf16*)(ws + take((size_t)3*DM*DM*2));       // Wq|Wk|Wv transposed bf16
  __bf16* WoT = (__bf16*)(ws + take((size_t)DM*DM*2));         // Wo transposed bf16
  __bf16* xb  = (__bf16*)(ws + take((size_t)NROWS*DM*2));      // x bf16
  float*  qkv = (float*) (ws + take((size_t)NROWS*3*DM*4));    // q|k|v f32, row stride 3*DM
  __bf16* qbb = (__bf16*)(ws + take((size_t)NROWS*DM*2));      // rmsnorm(q) bf16
  __bf16* kbb = (__bf16*)(ws + take((size_t)NROWS*DM*2));      // quant(rmsnorm(k)) bf16
  __bf16* vtb = (__bf16*)(ws + take((size_t)NROWS*DM*2));      // quant(v) bf16, (B,H,Dh,S)
  __bf16* obb = (__bf16*)(ws + take((size_t)NROWS*DM*2));      // attention out bf16

  // 1) prep: weight transpose+convert, x convert
  k_wtrans<<<dim3(DM/32, DM/32, 4), dim3(32,8), 0, stream>>>(Wq, Wk, Wv, Wo, WT3, WoT);
  k_cvt<<<(NROWS*DM/4 + 255)/256, 256, 0, stream>>>(x, xb, NROWS*DM/4);

  // 2) fused QKV projection: (2048 x 2048) @ (2048 x 6144)
  k_gemm<<<dim3(3*DM/64, NROWS/128), 256, 0, stream>>>(xb, WT3, qkv, NROWS, 3*DM, DM);

  // 3) rmsnorm: q -> bf16 directly; k -> f32 in place (quant reads it)
  k_rmsnorm<<<NROWS, 256, 0, stream>>>(qkv + 0,  qn_w, nullptr,  qbb,     3*DM, 0);
  k_rmsnorm<<<NROWS, 256, 0, stream>>>(qkv + DM, kn_w, qkv + DM, nullptr, 3*DM, 3*DM);

  // 4) RotorQuant roundtrip: k row-major bf16, v transposed bf16
  k_quant<<<(BATCH*HEADS*SEQ)/256, 256, 0, stream>>>(qkv + DM,   3*DM, Mk, Mtk, Ck, kbb, nullptr);
  k_quant<<<(BATCH*HEADS*SEQ)/256, 256, 0, stream>>>(qkv + 2*DM, 3*DM, Mv, Mtv, Cv, nullptr, vtb);

  // 5) flash attention
  k_attn<<<dim3(BATCH*HEADS, SEQ/128), 256, 0, stream>>>(qbb, kbb, vtb, obb);

  // 6) output projection -> d_out f32
  k_gemm<<<dim3(DM/64, NROWS/128), 256, 0, stream>>>(obb, WoT, (float*)d_out, NROWS, DM, DM);
}
